// R2RSSM_36438502539529
// MI455X (gfx1250) — matrix operand order repaired and checked by reading, not compiled, not run on hardware
//
#include <hip/hip_runtime.h>
#include <hip/hip_bf16.h>

// ---------------------------------------------------------------------------
// CDNA5 (gfx1250, wave32) RSSM step using V_WMMA_F32_16X16X32_BF16.
// All GEMM operands bf16, accumulation f32. Norm/GRU/argmax math in f32.
// Round 2: compile-time K/LDA (no per-iter concat select), unrolled K loops.
// ---------------------------------------------------------------------------

typedef __bf16 bf16_t;
typedef __attribute__((ext_vector_type(16))) __bf16 v16bf;
typedef __attribute__((ext_vector_type(8)))  __bf16 v8bf;
typedef __attribute__((ext_vector_type(8)))  float  v8f;

#define RSSM_B   4096
#define RSSM_EPS 1e-4f

// ---- WMMA fragment helpers (layouts per CDNA5 ISA §7.12.2) -----

// A 16x32 bf16 fragment. rowptr = &A[row][k0] (row-major bf16). h = lane>>4.
// lane half h holds K = [8h, 8h+8) in elements 0..7 and [16+8h, 16+8h+8) in 8..15.
__device__ __forceinline__ v16bf load_a_frag(const bf16_t* __restrict__ rowptr, int h) {
  v8bf lo = *(const v8bf*)(rowptr + h * 8);
  v8bf hi = *(const v8bf*)(rowptr + 16 + h * 8);
  v16bf a;
#pragma unroll
  for (int i = 0; i < 8; ++i) { a[i] = lo[i]; a[i + 8] = hi[i]; }
  return a;
}

// B 32x16 bf16 fragment from transposed weights Wt[out][in].
// wrow = &Wt[ncol][k0]; lane half hb holds K = [8hb, 8hb+8) in 0..7 and [16+8hb, 16+8hb+8) in 8..15.
__device__ __forceinline__ v16bf load_b_frag(const bf16_t* __restrict__ wrow, int hb) {
  v8bf lo = *(const v8bf*)(wrow + hb * 8);
  v8bf hi = *(const v8bf*)(wrow + 16 + hb * 8);
  v16bf b;
#pragma unroll
  for (int i = 0; i < 8; ++i) { b[i] = lo[i]; b[i + 8] = hi[i]; }
  return b;
}

__device__ __forceinline__ v8f wmma_bf16(v16bf a, v16bf b, v8f c) {
  // (neg_a, A, neg_b, B, c_mod, C, reuse_a, reuse_b)
  return __builtin_amdgcn_wmma_f32_16x16x32_bf16(false, a, false, b, (short)0, c,
                                                 false, false);
}

__device__ __forceinline__ float sigmoid_f(float x) { return 1.f / (1.f + __expf(-x)); }
__device__ __forceinline__ float silu_f(float x)    { return x / (1.f + __expf(-x)); }

// ------------------------------- prep kernels ------------------------------

__global__ void k_f32_to_bf16(const float* __restrict__ src, bf16_t* __restrict__ dst, int n) {
  int i = blockIdx.x * blockDim.x + threadIdx.x;
  if (i < n) dst[i] = (bf16_t)src[i];
}

// action (B,17) -> normalized, zero-padded bf16 (B,32)
__global__ void k_act_prep(const float* __restrict__ act, bf16_t* __restrict__ dst) {
  int i = blockIdx.x * blockDim.x + threadIdx.x;
  if (i >= RSSM_B * 32) return;
  int r = i >> 5, j = i & 31;
  float v = 0.f;
  if (j < 17) { float a = act[r * 17 + j]; v = a / fmaxf(fabsf(a), 1.f); }
  dst[i] = (bf16_t)v;
}

// src (K,N) f32 row-major -> dst (N,Kp) bf16 (K padded with zeros to Kp)
__global__ void k_transpose_w(const float* __restrict__ src, bf16_t* __restrict__ dst,
                              int K, int N, int Kp) {
  int i = blockIdx.x * blockDim.x + threadIdx.x;
  if (i >= N * Kp) return;
  int n = i / Kp, k = i % Kp;
  dst[i] = (k < K) ? (bf16_t)src[(size_t)k * N + n] : (bf16_t)0.f;
}

// block kernel src (opb, ipb, 8) f32 -> dst [g][o][i] bf16
__global__ void k_prep_block(const float* __restrict__ src, bf16_t* __restrict__ dst,
                             int opb, int ipb) {
  int i = blockIdx.x * blockDim.x + threadIdx.x;
  int per = opb * ipb;
  if (i >= per * 8) return;
  int g = i / per, rem = i % per;
  int o = rem / ipb, ii = rem % ipb;
  dst[i] = (bf16_t)src[((size_t)o * ipb + ii) * 8 + g];
}

// --------------------- fused GEMM + bias + rmsnorm + silu ------------------
// N fixed at 256. 16 waves: wave w owns N-tile w. Then wave r normalizes row r.
// A is a 2-segment concat: k < KSPLIT -> A0 (LDA0), else A1 (LDA1).
// K/KSPLIT/LDA are compile-time so the K loop has immediate-offset addressing.
template <int K, int KSPLIT, int LDA0, int LDA1>
__global__ __launch_bounds__(512)
void k_gemm_norm_silu(const bf16_t* __restrict__ A0,
                      const bf16_t* __restrict__ A1,
                      const bf16_t* __restrict__ Wt,
                      const float* __restrict__ bias,
                      const float* __restrict__ scale,
                      bf16_t* __restrict__ out, int ldo, int ocol) {
  __shared__ float sv[16][256];
  const int tid = threadIdx.x, lane = tid & 31, wave = tid >> 5;
  const int row0 = blockIdx.x * 16;
  const int m = lane & 15, h = lane >> 4;

  v8f acc = {};
  const bf16_t* wp = Wt + (size_t)(wave * 16 + m) * K;
  {
    const bf16_t* ap = A0 + (size_t)(row0 + m) * LDA0;
#pragma unroll 4
    for (int k0 = 0; k0 < KSPLIT; k0 += 32)
      acc = wmma_bf16(load_a_frag(ap + k0, h), load_b_frag(wp + k0, h), acc);
  }
  if constexpr (KSPLIT < K) {
    const bf16_t* ap = A1 + (size_t)(row0 + m) * LDA1;
#pragma unroll 4
    for (int k0 = KSPLIT; k0 < K; k0 += 32)
      acc = wmma_bf16(load_a_frag(ap + (k0 - KSPLIT), h), load_b_frag(wp + k0, h), acc);
  }
  {
    const int n = lane & 15, hc = lane >> 4;
    float bv = bias[wave * 16 + n];
#pragma unroll
    for (int v = 0; v < 8; ++v) sv[v + 8 * hc][wave * 16 + n] = acc[v] + bv;
  }
  __syncthreads();
  {
    const int r = wave;  // wave r handles row r (16 waves, 16 rows)
    float vals[8], s = 0.f;
#pragma unroll
    for (int i = 0; i < 8; ++i) { float v = sv[r][lane * 8 + i]; vals[i] = v; s += v * v; }
#pragma unroll
    for (int off = 16; off > 0; off >>= 1) s += __shfl_xor(s, off, 32);
    float invr = rsqrtf(s * (1.f / 256.f) + RSSM_EPS);
    bf16_t* orow = out + (size_t)(row0 + r) * ldo + ocol;
#pragma unroll
    for (int i = 0; i < 8; ++i) {
      int c = lane * 8 + i;
      orow[c] = (bf16_t)silu_f(vals[i] * invr * scale[c]);
    }
  }
}

// -------- hid0 block-linear: A = [deter_block(256) | xcat(768)], K=1024 ----
// grid (B/16, 2, 8 blocks); 8 waves; raw f32 (+bias) out (norm spans blocks).
__global__ __launch_bounds__(256)
void k_hid0_gemm(const bf16_t* __restrict__ deter_bf, const bf16_t* __restrict__ xcat,
                 const bf16_t* __restrict__ Wt, const float* __restrict__ bias,
                 float* __restrict__ raw) {
  const int tid = threadIdx.x, lane = tid & 31, wave = tid >> 5;
  const int row0 = blockIdx.x * 16;
  const int g = blockIdx.z;
  const int nt = blockIdx.y * 8 + wave;  // 0..15
  const int m = lane & 15, h = lane >> 4;

  const bf16_t* wp = Wt + ((size_t)g * 256 + nt * 16 + m) * 1024;
  const bf16_t* dp = deter_bf + (size_t)(row0 + m) * 2048 + g * 256;
  const bf16_t* xp = xcat + (size_t)(row0 + m) * 768;
  v8f acc = {};
#pragma unroll 4
  for (int k0 = 0; k0 < 256; k0 += 32)      // deter-block segment
    acc = wmma_bf16(load_a_frag(dp + k0, h), load_b_frag(wp + k0, h), acc);
#pragma unroll 4
  for (int k0 = 256; k0 < 1024; k0 += 32)   // shared-x segment
    acc = wmma_bf16(load_a_frag(xp + (k0 - 256), h), load_b_frag(wp + k0, h), acc);
  {
    const int n = lane & 15, hc = lane >> 4;
    const int col = g * 256 + nt * 16 + n;
    float bv = bias[col];
#pragma unroll
    for (int v = 0; v < 8; ++v)
      raw[(size_t)(row0 + v + 8 * hc) * 2048 + col] = acc[v] + bv;
  }
}

// rmsnorm+silu over full 2048-wide row -> bf16
__global__ __launch_bounds__(256)
void k_norm_silu_2048(const float* __restrict__ raw, const float* __restrict__ scale,
                      bf16_t* __restrict__ out) {
  __shared__ float red[8];
  const int row = blockIdx.x, tid = threadIdx.x, lane = tid & 31, wave = tid >> 5;
  const float* r = raw + (size_t)row * 2048;
  float s = 0.f;
#pragma unroll
  for (int c = tid; c < 2048; c += 256) { float v = r[c]; s += v * v; }
#pragma unroll
  for (int off = 16; off > 0; off >>= 1) s += __shfl_xor(s, off, 32);
  if (lane == 0) red[wave] = s;
  __syncthreads();
  float tot = 0.f;
#pragma unroll
  for (int i = 0; i < 8; ++i) tot += red[i];
  float invr = rsqrtf(tot * (1.f / 2048.f) + RSSM_EPS);
  bf16_t* o = out + (size_t)row * 2048;
#pragma unroll
  for (int c = tid; c < 2048; c += 256)
    o[c] = (bf16_t)silu_f(r[c] * invr * scale[c]);
}

// ------------- GRU block-linear (K=256, N=768) + fused gate math -----------
// grid (B/16, 8 blocks); 8 waves, 6 N-tiles each; 16x768 gates staged in LDS.
__global__ __launch_bounds__(256)
void k_gru(const bf16_t* __restrict__ xh, const bf16_t* __restrict__ Wt,
           const float* __restrict__ bias, const float* __restrict__ deter,
           float* __restrict__ nd_out, bf16_t* __restrict__ nd_bf) {
  __shared__ float gate[16][768];  // 48 KB
  const int tid = threadIdx.x, lane = tid & 31, wave = tid >> 5;
  const int row0 = blockIdx.x * 16, g = blockIdx.y;
  const int m = lane & 15, h = lane >> 4;

  v8f acc[6];
#pragma unroll
  for (int t = 0; t < 6; ++t) acc[t] = (v8f){};
  const bf16_t* ap = xh + (size_t)(row0 + m) * 2048 + g * 256;
  const bf16_t* wp = Wt + ((size_t)g * 768 + wave * 16 + m) * 256;  // +t*128*256
#pragma unroll 2
  for (int k0 = 0; k0 < 256; k0 += 32) {
    v16bf a = load_a_frag(ap + k0, h);  // shared across the 6 tiles
#pragma unroll
    for (int t = 0; t < 6; ++t) {
      v16bf b = load_b_frag(wp + (size_t)t * 128 * 256 + k0, h);
      acc[t] = wmma_bf16(a, b, acc[t]);
    }
  }
  {
    const int n = lane & 15, hc = lane >> 4;
#pragma unroll
    for (int t = 0; t < 6; ++t) {
      const int col = (wave + 8 * t) * 16 + n;  // 0..767
      float bv = bias[g * 768 + col];
#pragma unroll
      for (int v = 0; v < 8; ++v) gate[v + 8 * hc][col] = acc[t][v] + bv;
    }
  }
  __syncthreads();
#pragma unroll
  for (int idx = tid; idx < 16 * 256; idx += 256) {
    const int r = idx >> 8, j = idx & 255;
    float g0 = gate[r][j], g1 = gate[r][256 + j], g2 = gate[r][512 + j];
    float reset  = sigmoid_f(g0);
    float update = sigmoid_f(g2 - 1.f);
    size_t o = (size_t)(row0 + r) * 2048 + g * 256 + j;
    float nd = update * tanhf(reset * g1) + (1.f - update) * deter[o];
    nd_out[o] = nd;
    nd_bf[o]  = (bf16_t)nd;
  }
}

// ------- obs_out GEMM (K=256, N=512) + logit store + one-hot argmax --------
// Each 16x16 tile is exactly one categorical group (SD=16): argmax across the
// 16 lanes of each half-wave (rows differ by half), first-max tie rule.
__global__ __launch_bounds__(256)
void k_obs_out(const bf16_t* __restrict__ hbf, const bf16_t* __restrict__ Wt,
               const float* __restrict__ bias, float* __restrict__ logit_out,
               float* __restrict__ stoch_out) {
  const int tid = threadIdx.x, lane = tid & 31, wave = tid >> 5;
  const int row0 = blockIdx.x * 16;
  const int nt = blockIdx.y * 8 + wave;  // group 0..31
  const int m = lane & 15, h = lane >> 4;

  v8f acc = {};
  const bf16_t* ap = hbf + (size_t)(row0 + m) * 256;
  const bf16_t* wp = Wt + (size_t)(nt * 16 + m) * 256;
#pragma unroll 4
  for (int k0 = 0; k0 < 256; k0 += 32)
    acc = wmma_bf16(load_a_frag(ap + k0, h), load_b_frag(wp + k0, h), acc);

  const int n = lane & 15, hc = lane >> 4;
  const float bv = bias[nt * 16 + n];
#pragma unroll
  for (int v = 0; v < 8; ++v) {
    const int row = row0 + v + 8 * hc;
    float val = acc[v] + bv;
    logit_out[(size_t)row * 512 + nt * 16 + n] = val;
    float bvv = val; int bi = n;
#pragma unroll
    for (int off = 1; off < 16; off <<= 1) {  // butterfly within each 16-lane half
      float ov = __shfl_xor(bvv, off, 32);
      int   oi = __shfl_xor(bi, off, 32);
      if (ov > bvv || (ov == bvv && oi < bi)) { bvv = ov; bi = oi; }
    }
    stoch_out[(size_t)row * 512 + nt * 16 + n] = (n == bi) ? 1.f : 0.f;
  }
}

// ------------------------------- launcher ----------------------------------

extern "C" void kernel_launch(void* const* d_in, const int* in_sizes, int n_in,
                              void* d_out, int out_size, void* d_ws, size_t ws_size,
                              hipStream_t stream) {
  (void)in_sizes; (void)n_in; (void)out_size; (void)ws_size;
  constexpr int B = RSSM_B;

  const float* stoch   = (const float*)d_in[0];
  const float* deter   = (const float*)d_in[1];
  const float* action  = (const float*)d_in[2];
  const float* embed   = (const float*)d_in[3];
  const float* in0_w   = (const float*)d_in[4];
  const float* in0_b   = (const float*)d_in[5];
  const float* n0s     = (const float*)d_in[6];
  const float* in1_w   = (const float*)d_in[7];
  const float* in1_b   = (const float*)d_in[8];
  const float* n1s     = (const float*)d_in[9];
  const float* in2_w   = (const float*)d_in[10];
  const float* in2_b   = (const float*)d_in[11];
  const float* n2s     = (const float*)d_in[12];
  const float* hid0_k  = (const float*)d_in[13];
  const float* hid0_b  = (const float*)d_in[14];
  const float* hn0s    = (const float*)d_in[15];
  const float* gru_k   = (const float*)d_in[16];
  const float* gru_b   = (const float*)d_in[17];
  const float* obs0_w  = (const float*)d_in[18];
  const float* obs0_b  = (const float*)d_in[19];
  const float* obs0_s  = (const float*)d_in[20];
  const float* obsot_w = (const float*)d_in[21];
  const float* obsot_b = (const float*)d_in[22];

  float* out       = (float*)d_out;
  float* out_stoch = out;                                       // (B,32,16)
  float* out_deter = out + (size_t)B * 512;                     // (B,2048)
  float* out_logit = out + (size_t)B * 512 + (size_t)B * 2048;  // (B,32,16)

  // workspace carve-out (~115 MB), 256B aligned
  size_t off = 0;
  auto walloc = [&](size_t elems) -> bf16_t* {
    bf16_t* p = (bf16_t*)((char*)d_ws + off);
    off += ((elems * sizeof(bf16_t)) + 255) & ~(size_t)255;
    return p;
  };
  bf16_t* w_in0t   = walloc((size_t)256 * 2048);
  bf16_t* w_in1t   = walloc((size_t)256 * 512);
  bf16_t* w_in2t   = walloc((size_t)256 * 32);
  bf16_t* w_hid0t  = walloc((size_t)8 * 256 * 1024);
  bf16_t* w_grut   = walloc((size_t)8 * 768 * 256);
  bf16_t* w_obs0t  = walloc((size_t)256 * 3072);
  bf16_t* w_obsot  = walloc((size_t)512 * 256);
  bf16_t* deter_bf = walloc((size_t)B * 2048);
  bf16_t* stoch_bf = walloc((size_t)B * 512);
  bf16_t* act_bf   = walloc((size_t)B * 32);
  bf16_t* embed_bf = walloc((size_t)B * 1024);
  bf16_t* xcat     = walloc((size_t)B * 768);
  bf16_t* xh_bf    = walloc((size_t)B * 2048);
  bf16_t* ndet_bf  = walloc((size_t)B * 2048);
  bf16_t* h_bf     = walloc((size_t)B * 256);
  float*  hid_raw  = (float*)((char*)d_ws + off);
  off += (size_t)B * 2048 * sizeof(float);

  auto blks = [](long long n, int t) { return dim3((unsigned)((n + t - 1) / t)); };

  // ---- prep: bf16 conversions + weight transposes ----
  k_f32_to_bf16<<<blks((long long)B * 2048, 256), 256, 0, stream>>>(deter, deter_bf, B * 2048);
  k_f32_to_bf16<<<blks((long long)B * 512, 256), 256, 0, stream>>>(stoch, stoch_bf, B * 512);
  k_f32_to_bf16<<<blks((long long)B * 1024, 256), 256, 0, stream>>>(embed, embed_bf, B * 1024);
  k_act_prep<<<blks((long long)B * 32, 256), 256, 0, stream>>>(action, act_bf);
  k_transpose_w<<<blks(256LL * 2048, 256), 256, 0, stream>>>(in0_w, w_in0t, 2048, 256, 2048);
  k_transpose_w<<<blks(256LL * 512, 256), 256, 0, stream>>>(in1_w, w_in1t, 512, 256, 512);
  k_transpose_w<<<blks(256LL * 32, 256), 256, 0, stream>>>(in2_w, w_in2t, 17, 256, 32);
  k_transpose_w<<<blks(256LL * 3072, 256), 256, 0, stream>>>(obs0_w, w_obs0t, 3072, 256, 3072);
  k_transpose_w<<<blks(512LL * 256, 256), 256, 0, stream>>>(obsot_w, w_obsot, 256, 512, 256);
  k_prep_block<<<blks(8LL * 256 * 1024, 256), 256, 0, stream>>>(hid0_k, w_hid0t, 256, 1024);
  k_prep_block<<<blks(8LL * 768 * 256, 256), 256, 0, stream>>>(gru_k, w_grut, 768, 256);

  // ---- input nets: x = [silu(rms(deter@W0)) | silu(rms(stoch@W1)) | silu(rms(act@W2))] ----
  k_gemm_norm_silu<2048, 2048, 2048, 2048>
      <<<dim3(B / 16), 512, 0, stream>>>(deter_bf, deter_bf, w_in0t, in0_b, n0s, xcat, 768, 0);
  k_gemm_norm_silu<512, 512, 512, 512>
      <<<dim3(B / 16), 512, 0, stream>>>(stoch_bf, stoch_bf, w_in1t, in1_b, n1s, xcat, 768, 256);
  k_gemm_norm_silu<32, 32, 32, 32>
      <<<dim3(B / 16), 512, 0, stream>>>(act_bf, act_bf, w_in2t, in2_b, n2s, xcat, 768, 512);

  // ---- hid0 block-linear + 2048-wide rmsnorm/silu ----
  k_hid0_gemm<<<dim3(B / 16, 2, 8), 256, 0, stream>>>(deter_bf, xcat, w_hid0t, hid0_b, hid_raw);
  k_norm_silu_2048<<<dim3(B), 256, 0, stream>>>(hid_raw, hn0s, xh_bf);

  // ---- GRU gates + fused update -> new_deter ----
  k_gru<<<dim3(B / 16, 8), 256, 0, stream>>>(xh_bf, w_grut, gru_b, deter, out_deter, ndet_bf);

  // ---- posterior head: h = silu(rms([new_deter|embed] @ W)) ; logit/one-hot ----
  k_gemm_norm_silu<3072, 2048, 2048, 1024>
      <<<dim3(B / 16), 512, 0, stream>>>(ndet_bf, embed_bf, w_obs0t, obs0_b, obs0_s, h_bf, 256, 0);
  k_obs_out<<<dim3(B / 16, 4), 256, 0, stream>>>(h_bf, w_obsot, obsot_b, out_logit, out_stoch);
}
